// QChunk_8272107012156
// MI455X (gfx1250) — matrix operand order repaired and checked by reading, not compiled, not run on hardware
//
#include <hip/hip_runtime.h>
#include <type_traits>

// Sliding-window causal attention (window=1024 incl. self), GQA 16Q/8KV heads,
// d=128, fp32 I/O, bf16 WMMA compute with fp32 accumulation.
// Softmax in exp2 domain (Q pre-scaled by log2 e); row-sums via WMMA vs ones.
// Per-wave tile classification: skip fully-masked tiles, maskless fast path
// for fully-open tiles, predicated path only on boundary tiles.

#define TLEN      4096
#define NHEADS    16
#define NKVH      8
#define DHEAD     128
#define WIN       1024
#define QROWS_WG  128
#define QROWS_WV  16
#define KV_STEP   64
#define LOG2E     1.4426950408889634f

typedef __attribute__((ext_vector_type(16))) __bf16 v16bf;
typedef __attribute__((ext_vector_type(8)))  __bf16 v8bf;
typedef __attribute__((ext_vector_type(8)))  float  v8f;
typedef __attribute__((ext_vector_type(4)))  float  v4f;

union FragBF {
    v16bf bf;
    v8bf  v8[2];
    unsigned short u[16];
};
union Bf8 {
    v8bf v;
    unsigned short u[8];
};

__device__ __forceinline__ v8f loadv8(const float* p) {
    v4f a = *(const v4f*)p;
    v4f b = *(const v4f*)(p + 4);
    return __builtin_shufflevector(a, b, 0, 1, 2, 3, 4, 5, 6, 7);
}

__device__ __forceinline__ float fexp2(float x) {
#if __has_builtin(__builtin_amdgcn_exp2f)
    return __builtin_amdgcn_exp2f(x);
#else
    return exp2f(x);
#endif
}

__global__ __launch_bounds__(256)
void swa_fwd_kernel(const float* __restrict__ q,
                    const float* __restrict__ k,
                    const float* __restrict__ v,
                    float* __restrict__ out)
{
    __shared__ alignas(16) unsigned short lK [KV_STEP * DHEAD];         // [key][d]
    __shared__ alignas(16) unsigned short lVt[DHEAD * KV_STEP];         // [d][key]
    __shared__ alignas(16) unsigned short lP [8 * QROWS_WV * KV_STEP];  // per-wave P

    const int bq   = blockIdx.z;
    const int head = blockIdx.y;
    const int kvh  = head >> 1;
    const int q0   = blockIdx.x * QROWS_WG;

    const int tid  = threadIdx.x;
    const int wave = tid >> 5;
    const int lane = tid & 31;
    const int ln16 = lane & 15;
    const bool hi  = lane >= 16;
    const int kh   = hi ? 8 : 0;

    const int qw0 = q0 + wave * QROWS_WV;

    // ---- Q tile -> 4 bf16 A-fragments (16x32), pre-scaled by log2(e) ----
    FragBF aQ[4];
    {
        const float* qrow = q + (((size_t)bq * TLEN + (qw0 + ln16)) * NHEADS + head) * DHEAD;
        const int klo = hi ? 8 : 0;
#pragma unroll
        for (int j = 0; j < 4; ++j) {
            v8f t0 = loadv8(qrow + j * 32 + klo) * LOG2E;
            v8f t1 = loadv8(qrow + j * 32 + klo + 16) * LOG2E;
            aQ[j].v8[0] = __builtin_convertvector(t0, v8bf);
            aQ[j].v8[1] = __builtin_convertvector(t1, v8bf);
        }
    }

    // all-ones bf16 B fragment (row sums of P via WMMA)
    FragBF ones;
#pragma unroll
    for (int e = 0; e < 16; ++e) ones.u[e] = 0x3f80;

    v8f oAcc[8];
#pragma unroll
    for (int dt = 0; dt < 8; ++dt) oAcc[dt] = {};
    v8f lacc = {};
    float mrun[8];
#pragma unroll
    for (int i = 0; i < 8; ++i) mrun[i] = 0.0f;   // exp2(e - max(0,m)) -> 0 when masked

    int kstart = q0 - (WIN - 1);
    if (kstart < 0) kstart = 0;
    kstart &= ~(KV_STEP - 1);
    const int kend = q0 + QROWS_WG;

    unsigned short* pw = &lP[wave * (QROWS_WV * KV_STEP)];

    for (int kv0 = kstart; kv0 < kend; kv0 += KV_STEP) {
        __syncthreads();
        {   // cooperative K/V staging: fp32 -> bf16, V transposed, vectorized
            const int a2 = (tid & 31) * 2;        // key pair base
            const int dg = (tid >> 5) * 16;       // 16-wide d group
            const size_t b0 = (((size_t)bq * TLEN + (kv0 + a2)) * NKVH + kvh) * DHEAD + dg;
            const size_t b1 = b0 + (size_t)NKVH * DHEAD;
            *(v8bf*)&lK[a2 * DHEAD + dg]           = __builtin_convertvector(loadv8(k + b0),     v8bf);
            *(v8bf*)&lK[a2 * DHEAD + dg + 8]       = __builtin_convertvector(loadv8(k + b0 + 8), v8bf);
            *(v8bf*)&lK[(a2 + 1) * DHEAD + dg]     = __builtin_convertvector(loadv8(k + b1),     v8bf);
            *(v8bf*)&lK[(a2 + 1) * DHEAD + dg + 8] = __builtin_convertvector(loadv8(k + b1 + 8), v8bf);
            Bf8 c0a, c0b, c1a, c1b;
            c0a.v = __builtin_convertvector(loadv8(v + b0),     v8bf);
            c0b.v = __builtin_convertvector(loadv8(v + b0 + 8), v8bf);
            c1a.v = __builtin_convertvector(loadv8(v + b1),     v8bf);
            c1b.v = __builtin_convertvector(loadv8(v + b1 + 8), v8bf);
#pragma unroll
            for (int e = 0; e < 8; ++e) {
                *(unsigned int*)&lVt[(dg + e) * KV_STEP + a2] =
                    (unsigned int)c0a.u[e] | ((unsigned int)c1a.u[e] << 16);
                *(unsigned int*)&lVt[(dg + 8 + e) * KV_STEP + a2] =
                    (unsigned int)c0b.u[e] | ((unsigned int)c1b.u[e] << 16);
            }
        }
        __syncthreads();

        // ---- per-wave tile classification (wave-uniform, EXEC stays full) ----
        const int lastrow = qw0 + (QROWS_WV - 1);
        const bool fully_masked = (kv0 > lastrow) ||
                                  (kv0 + (KV_STEP - 1) < qw0 - (WIN - 1));
        if (fully_masked) continue;                 // only barriers for this wave
        const bool fully_open = (kv0 + (KV_STEP - 1) <= qw0) &&
                                (kv0 >= lastrow - (WIN - 1));

        auto step_body = [&](auto MASKC) {
            constexpr bool MASKED = decltype(MASKC)::value;

            // ---- S = Q K^T : 16 q x 64 keys, 16 WMMAs ----
            v8f s[4];
#pragma unroll
            for (int nt = 0; nt < 4; ++nt) {
                s[nt] = {};
#pragma unroll
                for (int j = 0; j < 4; ++j) {
                    FragBF bk;
                    const unsigned short* kp = &lK[(nt * 16 + ln16) * DHEAD + j * 32 + kh];
                    bk.v8[0] = *(const v8bf*)kp;
                    bk.v8[1] = *(const v8bf*)(kp + 16);
                    s[nt] = __builtin_amdgcn_wmma_f32_16x16x32_bf16(false, aQ[j].bf, false, bk.bf,
                                                                    (short)0, s[nt], false, false);
                }
            }

            // ---- online softmax (exp2 domain); emit P bf16 to LDS ----
            const int cb = kv0 + ln16;
#pragma unroll
            for (int i = 0; i < 8; ++i) {
                const int m  = i + (hi ? 8 : 0);
                const int pr = qw0 + m;
                float e[4];
#pragma unroll
                for (int nt = 0; nt < 4; ++nt) {
                    if constexpr (MASKED) {
                        const bool ok = (unsigned)(pr - (cb + nt * 16)) < (unsigned)WIN;
                        e[nt] = ok ? s[nt][i] : -3.0e38f;
                    } else {
                        e[nt] = s[nt][i];
                    }
                }
                float t = fmaxf(fmaxf(e[0], e[1]), fmaxf(e[2], e[3]));
#pragma unroll
                for (int w = 1; w <= 8; w <<= 1)
                    t = fmaxf(t, __shfl_xor(t, w, 32));
                const float mn    = fmaxf(mrun[i], t);
                const float alpha = fexp2(mrun[i] - mn);
                mrun[i] = mn;
                lacc[i] *= alpha;
#pragma unroll
                for (int dt = 0; dt < 8; ++dt)
                    oAcc[dt][i] *= alpha;
#pragma unroll
                for (int nt = 0; nt < 4; ++nt) {
                    const float pv = fexp2(e[nt] - mn);   // masked -> 0 naturally
                    pw[m * KV_STEP + nt * 16 + ln16] =
                        __builtin_bit_cast(unsigned short, (__bf16)pv);
                }
            }

            // ---- P fragments; row-sums via WMMA vs ones; O += P V ----
            FragBF aP0, aP1;
            {
                const int ko = hi ? 8 : 0;
                aP0.v8[0] = *(const v8bf*)&pw[ln16 * KV_STEP + ko];
                aP0.v8[1] = *(const v8bf*)&pw[ln16 * KV_STEP + 16 + ko];
                aP1.v8[0] = *(const v8bf*)&pw[ln16 * KV_STEP + 32 + ko];
                aP1.v8[1] = *(const v8bf*)&pw[ln16 * KV_STEP + 48 + ko];
            }
            lacc = __builtin_amdgcn_wmma_f32_16x16x32_bf16(false, aP0.bf, false, ones.bf,
                                                           (short)0, lacc, false, false);
            lacc = __builtin_amdgcn_wmma_f32_16x16x32_bf16(false, aP1.bf, false, ones.bf,
                                                           (short)0, lacc, false, false);
#pragma unroll
            for (int dt = 0; dt < 8; ++dt) {
                const unsigned short* vp = &lVt[(dt * 16 + ln16) * KV_STEP + kh];
                FragBF bv0, bv1;
                bv0.v8[0] = *(const v8bf*)vp;
                bv0.v8[1] = *(const v8bf*)(vp + 16);
                bv1.v8[0] = *(const v8bf*)(vp + 32);
                bv1.v8[1] = *(const v8bf*)(vp + 48);
                oAcc[dt] = __builtin_amdgcn_wmma_f32_16x16x32_bf16(false, aP0.bf, false, bv0.bf,
                                                                   (short)0, oAcc[dt], false, false);
                oAcc[dt] = __builtin_amdgcn_wmma_f32_16x16x32_bf16(false, aP1.bf, false, bv1.bf,
                                                                   (short)0, oAcc[dt], false, false);
            }
        };

        if (fully_open)
            step_body(std::false_type{});           // hot path: no mask VALU
        else
            step_body(std::true_type{});            // boundary tiles only
    }

    // ---- normalize and store ----
#pragma unroll
    for (int i = 0; i < 8; ++i) {
        const int m = i + (hi ? 8 : 0);
        const int p = qw0 + m;
        const float inv = 1.0f / lacc[i];
        float* orow = out + (((size_t)bq * TLEN + p) * NHEADS + head) * DHEAD;
#pragma unroll
        for (int dt = 0; dt < 8; ++dt)
            orow[dt * 16 + ln16] = oAcc[dt][i] * inv;
    }
}

extern "C" void kernel_launch(void* const* d_in, const int* in_sizes, int n_in,
                              void* d_out, int out_size, void* d_ws, size_t ws_size,
                              hipStream_t stream) {
    const float* q = (const float*)d_in[0];
    const float* k = (const float*)d_in[1];
    const float* v = (const float*)d_in[2];
    float* out = (float*)d_out;

    const int b = in_sizes[0] / (TLEN * NHEADS * DHEAD);   // 4
    dim3 grid(TLEN / QROWS_WG, NHEADS, b);
    dim3 block(256);
    swa_fwd_kernel<<<grid, block, 0, stream>>>(q, k, v, out);
}
